// MultiHeadSelfAttn_71116068487238
// MI455X (gfx1250) — matrix operand order repaired and checked by reading, not compiled, not run on hardware
//
#include <hip/hip_runtime.h>

// Problem constants (reference: B=2, S=2048, D=2048, H=16, HD=128)
#define Bv  2
#define Sv  2048
#define Dv  2048
#define Hv  16
#define HDv 128
#define Mv  (Bv * Sv)   // 4096 rows for the projection GEMMs

typedef __attribute__((ext_vector_type(16))) _Float16 v16h;  // WMMA A/B fragment (8 VGPRs)
typedef __attribute__((ext_vector_type(8)))  _Float16 v8h;   // 16-byte global load
typedef __attribute__((ext_vector_type(8)))  float    v8f;   // WMMA C/D fragment
typedef __attribute__((ext_vector_type(4)))  int      v4i;   // 16-byte async copy unit

// ---------------------------------------------------------------------------
// CDNA5 async global->LDS copy (ASYNCcnt path), with sync fallback
// Builtin signature: (v4i addrspace(1)* src, v4i addrspace(3)* dst, imm off, imm cpol)
// ---------------------------------------------------------------------------
#if defined(__has_builtin)
# if __has_builtin(__builtin_amdgcn_global_load_async_to_lds_b128) && \
     __has_builtin(__builtin_amdgcn_s_wait_asynccnt)
#  define USE_ASYNC_LDS 1
# endif
#endif

typedef __attribute__((address_space(1))) v4i v4i_as1;
typedef __attribute__((address_space(3))) v4i v4i_as3;

__device__ __forceinline__ void cp_b128(const _Float16* g, _Float16* l) {
#if defined(USE_ASYNC_LDS)
  __builtin_amdgcn_global_load_async_to_lds_b128((v4i_as1*)g, (v4i_as3*)l, 0, 0);
#else
  v8h d = *(const v8h*)g;
#pragma unroll
  for (int i = 0; i < 8; ++i) l[i] = d[i];
#endif
}

__device__ __forceinline__ void cp_wait() {
#if defined(USE_ASYNC_LDS)
  __builtin_amdgcn_s_wait_asynccnt(0);
#endif
}

__device__ __forceinline__ v8f zero_v8f() {
  v8f z;
#pragma unroll
  for (int i = 0; i < 8; ++i) z[i] = 0.0f;
  return z;
}

// A fragment (16xK tile, 16-bit): lane L<16 -> row L, K = kb+0..7 / kb+16..23 (kb=0)
//                                  lane L>=16 -> row L-16, kb=8. base at (row0, k0).
__device__ __forceinline__ v16h wmma_load_a(const _Float16* base, int ld) {
  const int lane = threadIdx.x & 31;
  const _Float16* p = base + (size_t)(lane & 15) * ld + ((lane >> 4) << 3);
  v16h f;
#pragma unroll
  for (int e = 0; e < 8; ++e) { f[e] = p[e]; f[e + 8] = p[e + 16]; }
  return f;
}

// Same pattern, two 16B vector loads (global Q; base+row*ld is 16B aligned)
__device__ __forceinline__ v16h wmma_load_a_vec(const _Float16* base, int ld) {
  const int lane = threadIdx.x & 31;
  const _Float16* p = base + (size_t)(lane & 15) * ld + ((lane >> 4) << 3);
  v8h d0 = *(const v8h*)p;
  v8h d1 = *(const v8h*)(p + 16);
  v16h f;
#pragma unroll
  for (int e = 0; e < 8; ++e) { f[e] = d0[e]; f[e + 8] = d1[e]; }
  return f;
}

// B fragment (Kx16 tile, 16-bit) from an [n][k] buffer:
// lane L<16 -> col L, K = 0..7 / 16..23 ; lane L>=16 -> col L-16, K = 8..15 / 24..31.
__device__ __forceinline__ v16h wmma_load_b(const _Float16* base, int ld) {
  const int lane = threadIdx.x & 31;
  const _Float16* p = base + (size_t)(lane & 15) * ld + ((lane >> 4) << 3);
  v16h f;
#pragma unroll
  for (int e = 0; e < 8; ++e) { f[e] = p[e]; f[e + 8] = p[e + 16]; }
  return f;
}

// ---------------------------------------------------------------------------
// fp32 -> f16 conversion (grid-stride)
// ---------------------------------------------------------------------------
__global__ void cvt_f32_to_f16(const float* __restrict__ src,
                               _Float16* __restrict__ dst, int n) {
  int i = blockIdx.x * blockDim.x + threadIdx.x;
  const int stride = gridDim.x * blockDim.x;
  for (; i < n; i += stride) dst[i] = (_Float16)src[i];
}

// ---------------------------------------------------------------------------
// NT GEMM:  C[m,n] = sum_k A[m,k] * B[n,k]   (y = x @ W.T)
//   A: Mv x Dv f16 row-major, B: Dv x Dv f16 row-major (z-strided by Dv*Dv)
//   mode 0: f16 output scattered to (B,H,S,HD), z-strided by Mv*Dv  (QKV)
//   mode 1: f32 output row-major Mv x Dv                            (final proj)
// 256 threads (8 waves), 128x128 tile; wave tile 64x32 (4x2 subtiles).
// Async double-buffered LDS staging: prefetch tile k+1 while computing tile k.
// ---------------------------------------------------------------------------
__global__ __launch_bounds__(256) void gemm_nt_wmma(
    const _Float16* __restrict__ A, const _Float16* __restrict__ Bm,
    void* __restrict__ Cbase, int mode) {
  __shared__ _Float16 lA[2][128][40];  // pad 8: rows stay 16B aligned for b128
  __shared__ _Float16 lB[2][128][40];

  const int t = threadIdx.x;
  const int wave = t >> 5;
  const int waveM = wave >> 2;  // 0..1
  const int waveN = wave & 3;   // 0..3
  const int rowBase = blockIdx.y * 128;
  const int colBase = blockIdx.x * 128;
  const _Float16* Bz = Bm + (size_t)blockIdx.z * (size_t)Dv * Dv;
  const size_t zC = (size_t)blockIdx.z * (size_t)Mv * Dv;

  v8f acc[4][2];
#pragma unroll
  for (int i = 0; i < 4; ++i)
#pragma unroll
    for (int j = 0; j < 2; ++j) acc[i][j] = zero_v8f();

  const int srow = t >> 1;        // 0..127
  const int sseg = (t & 1) * 16;  // 0 or 16
  const _Float16* Ag = A + (size_t)(rowBase + srow) * Dv + sseg;
  const _Float16* Bg = Bz + (size_t)(colBase + srow) * Dv + sseg;

  auto stage = [&](int buf, int kb) {
    cp_b128(Ag + kb,     &lA[buf][srow][sseg]);
    cp_b128(Ag + kb + 8, &lA[buf][srow][sseg + 8]);
    cp_b128(Bg + kb,     &lB[buf][srow][sseg]);
    cp_b128(Bg + kb + 8, &lB[buf][srow][sseg + 8]);
  };

  stage(0, 0);
  cp_wait();
  __syncthreads();

  int cur = 0;
  for (int kb = 0; kb < Dv; kb += 32) {
    const int nxt = cur ^ 1;
    if (kb + 32 < Dv) stage(nxt, kb + 32);  // overlap with WMMA below

    v16h af[4], bf[2];
#pragma unroll
    for (int ms = 0; ms < 4; ++ms)
      af[ms] = wmma_load_a(&lA[cur][waveM * 64 + ms * 16][0], 40);
#pragma unroll
    for (int ns = 0; ns < 2; ++ns)
      bf[ns] = wmma_load_b(&lB[cur][waveN * 32 + ns * 16][0], 40);
#pragma unroll
    for (int ms = 0; ms < 4; ++ms)
#pragma unroll
      for (int ns = 0; ns < 2; ++ns)
        acc[ms][ns] = __builtin_amdgcn_wmma_f32_16x16x32_f16(
            false, af[ms], false, bf[ns], (short)0, acc[ms][ns], false, false);

    cp_wait();
    __syncthreads();
    cur = nxt;
  }

  // Epilogue. C layout: element r -> M = r + (lane<16 ? 0 : 8), N = lane & 15.
  const int lane = t & 31;
  const int rb = (lane >> 4) << 3;
  const int nl = lane & 15;
#pragma unroll
  for (int ms = 0; ms < 4; ++ms)
#pragma unroll
    for (int ns = 0; ns < 2; ++ns)
#pragma unroll
      for (int r = 0; r < 8; ++r) {
        const int grow = rowBase + waveM * 64 + ms * 16 + rb + r;
        const int gcol = colBase + waveN * 32 + ns * 16 + nl;
        const float v = acc[ms][ns][r];
        if (mode == 0) {
          const int b = grow >> 11, s = grow & (Sv - 1);
          const int h = gcol >> 7, hd = gcol & (HDv - 1);
          _Float16* out = (_Float16*)Cbase + zC;
          out[(((size_t)(b * Hv + h)) * Sv + s) * HDv + hd] = (_Float16)v;
        } else {
          ((float*)Cbase)[(size_t)grow * Dv + gcol] = v;
        }
      }
}

// ---------------------------------------------------------------------------
// Flash attention, causal. Block = 8 waves x 16 q-rows = 128 q-rows per (b,h).
// Q/K/V in (B,H,S,HD) f16; output (B,S,D) f16.
// ---------------------------------------------------------------------------
__global__ __launch_bounds__(256) void attn_wmma(
    const _Float16* __restrict__ q, const _Float16* __restrict__ k,
    const _Float16* __restrict__ v, _Float16* __restrict__ o) {
  __shared__ _Float16 lK[32][136];    // K chunk, [key][hd] (async-staged, 16B rows)
  __shared__ _Float16 lVT[128][40];   // V chunk transposed, [hd][key]
  __shared__ _Float16 lP[8][16][34];  // per-wave P tile (C-layout -> A-layout bounce)

  const int t = threadIdx.x;
  const int wave = t >> 5;
  const int lane = t & 31;
  const int h = blockIdx.y;
  const int b = blockIdx.z;
  const int qb_blk = blockIdx.x * 128;
  const int qb = qb_blk + wave * 16;

  const size_t bh = ((size_t)b * Hv + h) * (size_t)Sv * HDv;
  const _Float16* qg = q + bh;
  const _Float16* kg = k + bh;
  const _Float16* vg = v + bh;

  // Resident Q fragments: 16 rows x 128 head-dim = 4 A fragments
  v16h aq[4];
#pragma unroll
  for (int kk = 0; kk < 4; ++kk)
    aq[kk] = wmma_load_a_vec(qg + (size_t)qb * HDv + kk * 32, HDv);

  v8f oacc[8];
#pragma unroll
  for (int j = 0; j < 8; ++j) oacc[j] = zero_v8f();
  float mrow[8], lrow[8];
#pragma unroll
  for (int r = 0; r < 8; ++r) { mrow[r] = -1e30f; lrow[r] = 0.0f; }

  const int rb = (lane >> 4) << 3;
  const int nl = lane & 15;
  const float scale = 0.08838834764831845f;  // HD^-0.5
  const int nchunks = (qb_blk >> 5) + 4;     // causal: keys <= qb_blk + 127

  const int skey = t >> 3;         // 0..31
  const int sseg = (t & 7) * 16;   // 0..112

  for (int c = 0; c < nchunks; ++c) {
    const int kc = c * 32;
    // Stage K row-major (async b128) and V transposed (sync)
    {
      const _Float16* ks = kg + (size_t)(kc + skey) * HDv + sseg;
      cp_b128(ks,     &lK[skey][sseg]);
      cp_b128(ks + 8, &lK[skey][sseg + 8]);
      const _Float16* vs = vg + (size_t)(kc + skey) * HDv + sseg;
      v8h e0 = *(const v8h*)vs;
      v8h e1 = *(const v8h*)(vs + 8);
#pragma unroll
      for (int i = 0; i < 8; ++i) { lVT[sseg + i][skey] = e0[i]; lVT[sseg + 8 + i][skey] = e1[i]; }
    }
    cp_wait();
    __syncthreads();

    if (kc <= qb + 15) {  // wave-uniform: at least one unmasked key for this wave
      // S = Q (16x128) @ K_chunk^T (128x32), f32 accumulate
      v8f s0 = zero_v8f(), s1 = zero_v8f();
#pragma unroll
      for (int kk = 0; kk < 4; ++kk) {
        v16h bk0 = wmma_load_b(&lK[0][kk * 32], 136);
        v16h bk1 = wmma_load_b(&lK[16][kk * 32], 136);
        s0 = __builtin_amdgcn_wmma_f32_16x16x32_f16(false, aq[kk], false, bk0, (short)0, s0, false, false);
        s1 = __builtin_amdgcn_wmma_f32_16x16x32_f16(false, aq[kk], false, bk1, (short)0, s1, false, false);
      }
      // Scale + causal mask (finite -1e30 to avoid inf-inf NaNs)
      float mx[8];
#pragma unroll
      for (int r = 0; r < 8; ++r) {
        const int qrow = qb + rb + r;
        s0[r] = (kc + nl <= qrow) ? s0[r] * scale : -1e30f;
        s1[r] = (kc + 16 + nl <= qrow) ? s1[r] * scale : -1e30f;
        mx[r] = fmaxf(s0[r], s1[r]);
      }
      // Row max across the 16 lanes holding each row
#pragma unroll
      for (int r = 0; r < 8; ++r)
#pragma unroll
        for (int m = 1; m <= 8; m <<= 1)
          mx[r] = fmaxf(mx[r], __shfl_xor(mx[r], m, 32));
      float al[8];
#pragma unroll
      for (int r = 0; r < 8; ++r) {
        const float mi = fmaxf(mrow[r], mx[r]);
        al[r] = __expf(mrow[r] - mi);
        mrow[r] = mi;
      }
      // P = exp(s - m): write to LDS in f16 (also get row sums)
      float rs[8];
#pragma unroll
      for (int r = 0; r < 8; ++r) {
        const float p0 = __expf(s0[r] - mrow[r]);
        const float p1 = __expf(s1[r] - mrow[r]);
        lP[wave][rb + r][nl] = (_Float16)p0;
        lP[wave][rb + r][16 + nl] = (_Float16)p1;
        rs[r] = p0 + p1;
      }
#pragma unroll
      for (int r = 0; r < 8; ++r) {
#pragma unroll
        for (int m = 1; m <= 8; m <<= 1) rs[r] += __shfl_xor(rs[r], m, 32);
        lrow[r] = lrow[r] * al[r] + rs[r];
      }
      // Rescale O by alpha per row
#pragma unroll
      for (int j = 0; j < 8; ++j)
#pragma unroll
        for (int r = 0; r < 8; ++r) oacc[j][r] *= al[r];
      // O += P (16x32) @ V_chunk (32x128)
      v16h ap = wmma_load_a(&lP[wave][0][0], 34);
#pragma unroll
      for (int j = 0; j < 8; ++j) {
        v16h bv = wmma_load_b(&lVT[j * 16][0], 40);
        oacc[j] = __builtin_amdgcn_wmma_f32_16x16x32_f16(false, ap, false, bv, (short)0, oacc[j], false, false);
      }
    }
    __syncthreads();
  }

  // Normalize and store to (B, S, D) f16
#pragma unroll
  for (int j = 0; j < 8; ++j)
#pragma unroll
    for (int r = 0; r < 8; ++r) {
      const float val = oacc[j][r] / lrow[r];
      const int qrow = qb + rb + r;
      const int col = h * HDv + j * 16 + nl;
      o[((size_t)b * Sv + qrow) * Dv + col] = (_Float16)val;
    }
}

// ---------------------------------------------------------------------------
// Host launcher
// ---------------------------------------------------------------------------
extern "C" void kernel_launch(void* const* d_in, const int* in_sizes, int n_in,
                              void* d_out, int out_size, void* d_ws, size_t ws_size,
                              hipStream_t stream) {
  (void)in_sizes; (void)n_in; (void)out_size; (void)ws_size;
  const float* x  = (const float*)d_in[0];
  // d_in[1]: causal mask (bool) — computed analytically in-kernel
  const float* Wq = (const float*)d_in[2];
  const float* Wk = (const float*)d_in[3];
  const float* Wv = (const float*)d_in[4];
  const float* Wo = (const float*)d_in[5];

  const size_t nBSD = (size_t)Bv * Sv * Dv;  // 8,388,608
  const size_t nDD  = (size_t)Dv * Dv;       // 4,194,304

  char* ws = (char*)d_ws;
  _Float16* x16   = (_Float16*)ws;  ws += nBSD * sizeof(_Float16);
  _Float16* w16   = (_Float16*)ws;  ws += 4 * nDD * sizeof(_Float16);   // Wq|Wk|Wv|Wo
  _Float16* qkv16 = (_Float16*)ws;  ws += 3 * nBSD * sizeof(_Float16);  // (3,B,H,S,HD)
  _Float16* a16   = (_Float16*)ws;                                      // (B,S,D)

  cvt_f32_to_f16<<<4096, 256, 0, stream>>>(x,  x16,           (int)nBSD);
  cvt_f32_to_f16<<<2048, 256, 0, stream>>>(Wq, w16 + 0 * nDD, (int)nDD);
  cvt_f32_to_f16<<<2048, 256, 0, stream>>>(Wk, w16 + 1 * nDD, (int)nDD);
  cvt_f32_to_f16<<<2048, 256, 0, stream>>>(Wv, w16 + 2 * nDD, (int)nDD);
  cvt_f32_to_f16<<<2048, 256, 0, stream>>>(Wo, w16 + 3 * nDD, (int)nDD);

  // QKV projections: z = 0/1/2 selects Wq/Wk/Wv, outputs into (B,H,S,HD)
  dim3 g0(Dv / 128, Mv / 128, 3);
  gemm_nt_wmma<<<g0, 256, 0, stream>>>(x16, w16, (void*)qkv16, 0);

  // Causal flash attention
  dim3 g1(Sv / 128, Hv, Bv);
  attn_wmma<<<g1, 256, 0, stream>>>(qkv16, qkv16 + nBSD, qkv16 + 2 * nBSD, a16);

  // Output projection -> fp32 (B,S,D)
  dim3 g2(Dv / 128, Mv / 128, 1);
  gemm_nt_wmma<<<g2, 256, 0, stream>>>(a16, w16 + 3 * nDD, d_out, 1);
}
